// MiniCPMV4_6ViTWindowAttentionMerger_83245056131921
// MI455X (gfx1250) — matrix operand order repaired and checked by reading, not compiled, not run on hardware
//
#include <hip/hip_runtime.h>
#include <cstdint>

// ---------------------------------------------------------------------------
// Types
// ---------------------------------------------------------------------------
typedef __bf16 v8bf   __attribute__((ext_vector_type(8)));
typedef __bf16 v16bf  __attribute__((ext_vector_type(16)));
typedef float  v8f    __attribute__((ext_vector_type(8)));
typedef int    v4i_   __attribute__((vector_size(4 * sizeof(int))));

#ifndef __has_builtin
#define __has_builtin(x) 0
#endif
#if __has_builtin(__builtin_amdgcn_global_load_async_to_lds_b128) && \
    __has_builtin(__builtin_amdgcn_s_wait_asynccnt)
#define USE_ASYNC_LDS 1
#else
#define USE_ASYNC_LDS 0
#endif

#define GLOBAL_V4(p) ((__attribute__((address_space(1))) v4i_*)(p))
#define LDS_V4(p)    ((__attribute__((address_space(3))) v4i_*)(p))

#define T_TOK   32768
#define NWIN    8192
#define DMODEL  768
#define NQKV    2304
#define D4      3072
#define I4DIM   12288

// ---------------------------------------------------------------------------
// Block-wide dual reduction (sum, sumsq), 256 threads = 8 wave32
// ---------------------------------------------------------------------------
__device__ __forceinline__ void block_sum2(float a, float b, float* sred,
                                           float& oa, float& ob) {
#pragma unroll
  for (int m = 16; m > 0; m >>= 1) {
    a += __shfl_xor(a, m, 32);
    b += __shfl_xor(b, m, 32);
  }
  int wid = threadIdx.x >> 5, lane = threadIdx.x & 31;
  if (lane == 0) { sred[wid] = a; sred[8 + wid] = b; }
  __syncthreads();
  if (threadIdx.x == 0) {
    float ta = 0.f, tb = 0.f;
    for (int k = 0; k < 8; ++k) { ta += sred[k]; tb += sred[8 + k]; }
    sred[16] = ta; sred[17] = tb;
  }
  __syncthreads();
  oa = sred[16]; ob = sred[17];
  __syncthreads();
}

// window-ordered token index -> original token index (B=8, H=W=64 fixed)
__device__ __forceinline__ size_t orig_of(int tw) {
  int w = tw >> 2, p = tw & 3;
  int b = w >> 10, rem = w & 1023, r = rem >> 5, c = rem & 31;
  return (size_t)b * 4096 + (size_t)(2 * r + (p >> 1)) * 64 + (size_t)(2 * c + (p & 1));
}

// ---------------------------------------------------------------------------
// f32 -> bf16 weight conversion
// ---------------------------------------------------------------------------
__global__ __launch_bounds__(256) void cvt_f32_bf16(const float* __restrict__ in,
                                                    __bf16* __restrict__ out, int n) {
  int i = blockIdx.x * 256 + threadIdx.x;
  if (i < n) out[i] = (__bf16)in[i];
}

__global__ void pack_bias3(const float* __restrict__ a, const float* __restrict__ b,
                           const float* __restrict__ c, float* __restrict__ out) {
  int i = blockIdx.x * 256 + threadIdx.x;
  if (i < DMODEL) out[i] = a[i];
  else if (i < 2 * DMODEL) out[i] = b[i - DMODEL];
  else if (i < 3 * DMODEL) out[i] = c[i - 2 * DMODEL];
}

// ---------------------------------------------------------------------------
// LayerNorm(768) + window gather + bf16 convert.  1 block per window token.
// ---------------------------------------------------------------------------
__global__ __launch_bounds__(256) void ln1_gather_kernel(
    const float* __restrict__ hs, const float* __restrict__ g,
    const float* __restrict__ b, __bf16* __restrict__ Xw) {
  __shared__ float sred[18];
  int tw = blockIdx.x, tid = threadIdx.x;
  const float* x = hs + orig_of(tw) * DMODEL;
  float v0 = x[tid], v1 = x[tid + 256], v2 = x[tid + 512];
  float ts, ts2;
  block_sum2(v0 + v1 + v2, v0 * v0 + v1 * v1 + v2 * v2, sred, ts, ts2);
  float mu  = ts * (1.f / 768.f);
  float inv = rsqrtf(ts2 * (1.f / 768.f) - mu * mu + 1e-6f);
  __bf16* o = Xw + (size_t)tw * DMODEL;
  o[tid]       = (__bf16)((v0 - mu) * inv * g[tid]       + b[tid]);
  o[tid + 256] = (__bf16)((v1 - mu) * inv * g[tid + 256] + b[tid + 256]);
  o[tid + 512] = (__bf16)((v2 - mu) * inv * g[tid + 512] + b[tid + 512]);
}

// ---------------------------------------------------------------------------
// bf16 WMMA GEMM: C[M,N] = epilogue(A[M,K] @ W[N,K]^T + bias)
//   EPI 0: -> bf16           EPI 1: GELU -> bf16
//   EPI 2: + res[orig(row)]  -> f32 (O-proj, residual gathered from original order)
//   EPI 3: + res[row]        -> f32 (final MLP output)
// Block tile 128x128x32; 8 waves, each wave 32x64 via 2x4 v_wmma_f32_16x16x32_bf16.
// Double-buffered LDS; tile k+1 streams via ASYNCcnt while tile k runs WMMA.
// Drain iteration peeled so the steady-state loop is branch-free.
// ---------------------------------------------------------------------------
#define BM 128
#define BN 128
#define BK 32
#define LDA 40
#define LDB 40

#if USE_ASYNC_LDS
__device__ __forceinline__ void issue_tile_copy(const __bf16* __restrict__ gA,
                                                const __bf16* __restrict__ gB,
                                                __bf16* bufA, __bf16* bufB,
                                                int K, int tid) {
#pragma unroll
  for (int t = 0; t < 2; ++t) {            // 512 16B chunks per tile, 2/thread
    int c = tid + t * 256;
    int r = c >> 2, off = (c & 3) * 8;
    __builtin_amdgcn_global_load_async_to_lds_b128(
        GLOBAL_V4(gA + (size_t)r * K + off), LDS_V4(&bufA[r * LDA + off]), 0, 0);
    __builtin_amdgcn_global_load_async_to_lds_b128(
        GLOBAL_V4(gB + (size_t)r * K + off), LDS_V4(&bufB[r * LDB + off]), 0, 0);
  }
}
#endif

// Fragment loads per ISA 16-bit layouts:
//  A 16x32: lanes 0-15 hold K{0..7,16..23} of row l15, lanes 16-31 K{8..15,24..31}
//  B 32x16: lane = N; lanes 0-15 K{0..7,16..23}, lanes 16-31 K{8..15,24..31}
__device__ __forceinline__ void mma_tile(const __bf16* __restrict__ tA,
                                         const __bf16* __restrict__ tB,
                                         v8f (&acc)[2][4],
                                         int wm, int wn, int half, int l15) {
  union Frag { v16bf v; v8bf h[2]; };
  Frag fa[2], fb[4];
#pragma unroll
  for (int i = 0; i < 2; ++i) {
    const __bf16* p = &tA[(wm * 32 + i * 16 + l15) * LDA + half * 8];
    fa[i].h[0] = *(const v8bf*)p;
    fa[i].h[1] = *(const v8bf*)(p + 16);
  }
#pragma unroll
  for (int j = 0; j < 4; ++j) {
    const __bf16* p = &tB[(wn * 64 + j * 16 + l15) * LDB + half * 8];
    fb[j].h[0] = *(const v8bf*)p;
    fb[j].h[1] = *(const v8bf*)(p + 16);
  }
#pragma unroll
  for (int i = 0; i < 2; ++i)
#pragma unroll
    for (int j = 0; j < 4; ++j)
      acc[i][j] = __builtin_amdgcn_wmma_f32_16x16x32_bf16(
          false, fa[i].v, false, fb[j].v, (short)0, acc[i][j], false, false);
}

template <int EPI>
__global__ __launch_bounds__(256) void gemm_bf16_kernel(
    const __bf16* __restrict__ A, const __bf16* __restrict__ W,
    const float* __restrict__ bias, const float* __restrict__ res,
    __bf16* __restrict__ outb, float* __restrict__ outf,
    int M, int N, int K) {
  __shared__ __bf16 sA[2][BM * LDA];
  __shared__ __bf16 sB[2][BN * LDB];
  const int tid  = threadIdx.x;
  const int lane = tid & 31;
  const int wid  = tid >> 5;
  const int wm   = wid >> 1;   // 0..3  (wave row)
  const int wn   = wid & 1;    // 0..1  (wave col)
  const int m0   = blockIdx.y * BM;
  const int n0   = blockIdx.x * BN;
  const int half = lane >> 4;  // 0/1
  const int l15  = lane & 15;

  v8f acc[2][4] = {};

  const __bf16* Abase = A + (size_t)m0 * K;
  const __bf16* Bbase = W + (size_t)n0 * K;
  const int nk = K / BK;

#if USE_ASYNC_LDS
  // Prologue: stream tile 0 into buffer 0.
  issue_tile_copy(Abase, Bbase, sA[0], sB[0], K, tid);
  int kb = 0;
  // Steady state (branch-free): issue k+1, prefetch k+2, compute k.
  for (; kb < nk - 1; ++kb) {
    const int cur = kb & 1;
    issue_tile_copy(Abase + (size_t)(kb + 1) * BK, Bbase + (size_t)(kb + 1) * BK,
                    sA[cur ^ 1], sB[cur ^ 1], K, tid);
    // Speculative L2 prefetch of tile k+2 (past-end prefetch is dropped).
    __builtin_prefetch(Abase + (size_t)(kb + 2) * BK, 0, 1);
    __builtin_prefetch(Bbase + (size_t)(kb + 2) * BK, 0, 1);
    // Async loads complete in order: <=4 outstanding means tile kb is in LDS.
    __builtin_amdgcn_s_wait_asynccnt(4);
    __syncthreads();
    mma_tile(sA[cur], sB[cur], acc, wm, wn, half, l15);
    __syncthreads();                       // buffer `cur` reusable next issue
  }
  // Drain: last tile.
  __builtin_amdgcn_s_wait_asynccnt(0);
  __syncthreads();
  mma_tile(sA[kb & 1], sB[kb & 1], acc, wm, wn, half, l15);
#else
  for (int kb = 0; kb < nk; ++kb) {
    const __bf16* gA = Abase + (size_t)kb * BK;
    const __bf16* gB = Bbase + (size_t)kb * BK;
#pragma unroll
    for (int t = 0; t < 2; ++t) {
      int c = tid + t * 256;
      int r = c >> 2, off = (c & 3) * 8;
      *(v8bf*)&sA[0][r * LDA + off] = *(const v8bf*)(gA + (size_t)r * K + off);
      *(v8bf*)&sB[0][r * LDB + off] = *(const v8bf*)(gB + (size_t)r * K + off);
    }
    __syncthreads();
    mma_tile(sA[0], sB[0], acc, wm, wn, half, l15);
    __syncthreads();
  }
#endif

  // Epilogue: C/D layout — VGPR vi holds M = vi + 8*(lane>=16), N = lane&15
#pragma unroll
  for (int i = 0; i < 2; ++i) {
#pragma unroll
    for (int j = 0; j < 4; ++j) {
#pragma unroll
      for (int vi = 0; vi < 8; ++vi) {
        int row = m0 + wm * 32 + i * 16 + vi + half * 8;
        int col = n0 + wn * 64 + j * 16 + l15;
        float v = acc[i][j][vi] + bias[col];
        if (EPI == 1) {
          float x = v;
          float t = tanhf(0.7978845608028654f * (x + 0.044715f * x * x * x));
          v = 0.5f * x * (1.0f + t);
        }
        if (EPI == 0 || EPI == 1) {
          outb[(size_t)row * N + col] = (__bf16)v;
        } else if (EPI == 2) {
          v += res[orig_of(row) * (size_t)N + col];
          outf[(size_t)row * N + col] = v;
        } else {
          v += res[(size_t)row * N + col];
          outf[(size_t)row * N + col] = v;
        }
      }
    }
  }
}

// ---------------------------------------------------------------------------
// 4-token window attention over fused QKV [T, 2304], 12 heads, hd=64.
// 1 block per window; 16 lanes per head: lane (qi,ki) computes score[qi][ki],
// then the qi-row output slice d = ki*16..ki*16+15 via shfl-shared weights.
// ---------------------------------------------------------------------------
__global__ __launch_bounds__(256) void win_attn_kernel(
    const __bf16* __restrict__ QKV, __bf16* __restrict__ O) {
  int w = blockIdx.x, tid = threadIdx.x;
  int h = tid >> 4;
  if (h >= 12) return;
  int l = tid & 15, qi = l >> 2, ki = l & 3;
  const __bf16* qrow = QKV + (size_t)(4 * w + qi) * NQKV + h * 64;
  const __bf16* krow = QKV + (size_t)(4 * w + ki) * NQKV + DMODEL + h * 64;
  float s = 0.f;
#pragma unroll 16
  for (int d = 0; d < 64; ++d) s += (float)qrow[d] * (float)krow[d];
  s *= 0.125f;  // 64^-0.5
  float mx = fmaxf(s, __shfl_xor(s, 1, 32));
  mx = fmaxf(mx, __shfl_xor(mx, 2, 32));
  float e = __expf(s - mx);
  float den = e + __shfl_xor(e, 1, 32);
  den += __shfl_xor(den, 2, 32);
  float a = e / den;
  int base = (tid & 31) & ~3;
  float o[16];
#pragma unroll
  for (int d = 0; d < 16; ++d) o[d] = 0.f;
#pragma unroll
  for (int k = 0; k < 4; ++k) {
    float ak = __shfl(a, base + k, 32);
    const __bf16* vr = QKV + (size_t)(4 * w + k) * NQKV + 2 * DMODEL + h * 64 + ki * 16;
#pragma unroll
    for (int d = 0; d < 16; ++d) o[d] += ak * (float)vr[d];
  }
  __bf16* orow = O + (size_t)(4 * w + qi) * DMODEL + h * 64 + ki * 16;
#pragma unroll
  for (int d = 0; d < 16; ++d) orow[d] = (__bf16)o[d];
}

// ---------------------------------------------------------------------------
// 2x2 merge: mean residual + LayerNorm(3072) -> bf16 H4. 1 block per window.
// hsw is window-ordered so each window's 4x768 group is a contiguous 3072 vec.
// ---------------------------------------------------------------------------
__global__ __launch_bounds__(256) void merge_ln_kernel(
    const float* __restrict__ hsw, const float* __restrict__ g,
    const float* __restrict__ b, __bf16* __restrict__ H4,
    float* __restrict__ mres) {
  __shared__ float sred[18];
  int w = blockIdx.x, tid = threadIdx.x;
  const float* base = hsw + (size_t)w * D4;
  float x[12], s = 0.f, s2 = 0.f;
#pragma unroll
  for (int t = 0; t < 12; ++t) {
    x[t] = base[tid + t * 256];
    s += x[t]; s2 += x[t] * x[t];
  }
  float ts, ts2;
  block_sum2(s, s2, sred, ts, ts2);
  float mu  = ts * (1.f / 3072.f);
  float inv = rsqrtf(ts2 * (1.f / 3072.f) - mu * mu + 1e-6f);
#pragma unroll
  for (int t = 0; t < 12; ++t) {
    int e = tid + t * 256;
    H4[(size_t)w * D4 + e] = (__bf16)((x[t] - mu) * inv * g[e] + b[e]);
  }
#pragma unroll
  for (int t = 0; t < 3; ++t) {
    int d = tid + t * 256;
    mres[(size_t)w * DMODEL + d] =
        0.25f * (base[d] + base[768 + d] + base[1536 + d] + base[2304 + d]);
  }
}

// ---------------------------------------------------------------------------
// Tail metadata outputs (new_target_sizes i64, new_cu_seqlens i32, max i32)
// ---------------------------------------------------------------------------
__global__ void write_meta_kernel(float* out, int out_size) {
  if (threadIdx.x != 0 || blockIdx.x != 0) return;
  const size_t NH = (size_t)NWIN * DMODEL;
  if ((size_t)out_size < NH + 43) return;
  char* p = (char*)(out + NH);
  long long* ts = (long long*)p;
  for (int i = 0; i < 16; ++i) ts[i] = 32;
  int* cu = (int*)(p + 16 * 8);
  for (int i = 0; i < 9; ++i) cu[i] = i * 1024;
  *(int*)(p + 16 * 8 + 9 * 4) = 1024;
}

// ---------------------------------------------------------------------------
// Host launcher
// ---------------------------------------------------------------------------
extern "C" void kernel_launch(void* const* d_in, const int* in_sizes, int n_in,
                              void* d_out, int out_size, void* d_ws, size_t ws_size,
                              hipStream_t stream) {
  (void)in_sizes; (void)n_in; (void)ws_size;
  const float* hs   = (const float*)d_in[0];
  const float* ln1g = (const float*)d_in[1];
  const float* ln1b = (const float*)d_in[2];
  const float* wq   = (const float*)d_in[3];
  const float* bq   = (const float*)d_in[4];
  const float* wk   = (const float*)d_in[5];
  const float* bk   = (const float*)d_in[6];
  const float* wv   = (const float*)d_in[7];
  const float* bv   = (const float*)d_in[8];
  const float* wo   = (const float*)d_in[9];
  const float* bo   = (const float*)d_in[10];
  const float* preg = (const float*)d_in[11];
  const float* preb = (const float*)d_in[12];
  const float* w1   = (const float*)d_in[13];
  const float* b1   = (const float*)d_in[14];
  const float* w2   = (const float*)d_in[15];
  const float* b2   = (const float*)d_in[16];

  size_t off = 0;
  auto alloc = [&](size_t bytes) -> void* {
    void* p = (char*)d_ws + off;
    off += (bytes + 255) & ~(size_t)255;
    return p;
  };
  __bf16* wqkvb = (__bf16*)alloc((size_t)NQKV * DMODEL * 2);  // [2304, 768]
  float*  bqkv  = (float*) alloc((size_t)NQKV * 4);
  __bf16* wob   = (__bf16*)alloc((size_t)DMODEL * DMODEL * 2);
  __bf16* w1b   = (__bf16*)alloc((size_t)I4DIM * D4 * 2);
  __bf16* w2b   = (__bf16*)alloc((size_t)DMODEL * I4DIM * 2);
  __bf16* Xw    = (__bf16*)alloc((size_t)T_TOK * DMODEL * 2);
  __bf16* QKVb  = (__bf16*)alloc((size_t)T_TOK * NQKV * 2);
  __bf16* Ob    = (__bf16*)alloc((size_t)T_TOK * DMODEL * 2);
  float*  hsw   = (float*) alloc((size_t)T_TOK * DMODEL * 4);
  __bf16* H4    = (__bf16*)alloc((size_t)NWIN * D4 * 2);
  float*  mres  = (float*) alloc((size_t)NWIN * DMODEL * 4);
  __bf16* H1    = (__bf16*)alloc((size_t)NWIN * I4DIM * 2);

  const int nDD = DMODEL * DMODEL;
  cvt_f32_bf16<<<(nDD + 255) / 256, 256, 0, stream>>>(wq, wqkvb, nDD);
  cvt_f32_bf16<<<(nDD + 255) / 256, 256, 0, stream>>>(wk, wqkvb + (size_t)nDD, nDD);
  cvt_f32_bf16<<<(nDD + 255) / 256, 256, 0, stream>>>(wv, wqkvb + (size_t)2 * nDD, nDD);
  pack_bias3<<<9, 256, 0, stream>>>(bq, bk, bv, bqkv);
  cvt_f32_bf16<<<(nDD + 255) / 256, 256, 0, stream>>>(wo, wob, nDD);
  const int n1 = I4DIM * D4;
  cvt_f32_bf16<<<(n1 + 255) / 256, 256, 0, stream>>>(w1, w1b, n1);
  const int n2 = DMODEL * I4DIM;
  cvt_f32_bf16<<<(n2 + 255) / 256, 256, 0, stream>>>(w2, w2b, n2);

  ln1_gather_kernel<<<T_TOK, 256, 0, stream>>>(hs, ln1g, ln1b, Xw);

  // Fused QKV projection: [32768,768] @ [2304,768]^T
  dim3 gQKV(NQKV / BN, T_TOK / BM);
  gemm_bf16_kernel<0><<<gQKV, 256, 0, stream>>>(Xw, wqkvb, bqkv, nullptr, QKVb,
                                                nullptr, T_TOK, NQKV, DMODEL);

  win_attn_kernel<<<NWIN, 256, 0, stream>>>(QKVb, Ob);

  dim3 gO(DMODEL / BN, T_TOK / BM);
  gemm_bf16_kernel<2><<<gO, 256, 0, stream>>>(Ob, wob, bo, hs, nullptr, hsw,
                                              T_TOK, DMODEL, DMODEL);

  merge_ln_kernel<<<NWIN, 256, 0, stream>>>(hsw, preg, preb, H4, mres);

  dim3 gMLP1(I4DIM / BN, NWIN / BM);
  gemm_bf16_kernel<1><<<gMLP1, 256, 0, stream>>>(H4, w1b, b1, nullptr, H1, nullptr,
                                                 NWIN, I4DIM, D4);
  dim3 gMLP2(DMODEL / BN, NWIN / BM);
  gemm_bf16_kernel<3><<<gMLP2, 256, 0, stream>>>(H1, w2b, b2, mres, nullptr,
                                                 (float*)d_out, NWIN, DMODEL, I4DIM);

  write_meta_kernel<<<1, 32, 0, stream>>>((float*)d_out, out_size);
}
